// BaiLingAttention_88948772700705
// MI455X (gfx1250) — matrix operand order repaired and checked by reading, not compiled, not run on hardware
//
#include <hip/hip_runtime.h>

#define H_SIZE   2048
#define N_HEADS  16
#define N_KV     4
#define HEAD_DIM 128
#define QKV_N    3072   /* (16 + 2*4) * 128 */
#define SEQ      2048
#define BATCH    2
#define TOKENS   (BATCH * SEQ)

// (1/sqrt(128)) * log2(e): folded into q so softmax runs in base-2 domain.
#define QSCALE_LOG2E 0.12752039735972247f

typedef __attribute__((ext_vector_type(16))) __bf16 v16bf;
typedef __attribute__((ext_vector_type(8)))  float  v8f;

// ---------------------------------------------------------------------------
// Fragment-major LDS layouts (wave32, 16x16x32 bf16 WMMA), per ISA 7.12.2:
//  A (16x32 MxK) element (m,k): lane = (m&15) + 16*((k>>3)&1),
//                               half = (k&7) + 8*((k>>4)&1)  [Qs; As, Ps: k = 16*(lane>>4) + half]
//  B (32x16 KxN) (k,n): lane = (n&15) + 16*((k>>3)&1), half as A [Ks; Bs, Vs: k = 16*(lane>>4) + half]
//  Blocks of 512 bf16; a fragment load is lane*16 .. lane*16+15 (32B).
//  C (16x16 f32) element j: lanes 0-15 -> M=j, N=lane; lanes 16-31 -> M=j+8.
// ---------------------------------------------------------------------------

// Low 32 bits of a generic pointer to __shared__ == LDS byte address (ISA 10.2).
__device__ __forceinline__ unsigned ldsoff(const void* p) {
  return (unsigned)(unsigned long long)p;
}
// Async DMA: 16 bytes global -> LDS, per lane; tracked by ASYNCcnt.
__device__ __forceinline__ void async_b128(unsigned lds, const void* g) {
  asm volatile("global_load_async_to_lds_b128 %0, %1, off"
               :: "v"(lds), "v"((unsigned long long)g) : "memory");
}
#define WAIT_ASYNC0() asm volatile("s_wait_asynccnt 0x0" ::: "memory")
#define WAIT_ASYNC4() asm volatile("s_wait_asynccnt 0x4" ::: "memory")

__device__ __forceinline__ v16bf frag16(const __bf16* block, int lane) {
  return *(const v16bf*)(block + lane * 16);
}

// ---------------------------------------------------------------------------
// Prep: fp32 -> bf16 elementwise (for hidden states)
// ---------------------------------------------------------------------------
__global__ __launch_bounds__(256)
void cvt_bf16_kernel(const float* __restrict__ in, __bf16* __restrict__ out, int n) {
  int i = blockIdx.x * 256 + threadIdx.x;
  if (i < n) out[i] = (__bf16)in[i];
}

// ---------------------------------------------------------------------------
// Prep: fp32 weight (K x N) -> bf16 in B-fragment-major global layout:
//   block id = (k>>5)*(N/16) + (n>>4); within block:
//   pos = ((n&15) + 16*((k>>4)&1))*16 + (k&15)
// ---------------------------------------------------------------------------
__global__ __launch_bounds__(256)
void wswz_kernel(const float* __restrict__ w, __bf16* __restrict__ o, int K, int N) {
  int t = blockIdx.x * 256 + threadIdx.x;      // one thread per (k, 16-col group)
  int nb16 = N >> 4;
  int k = t / nb16, nb = t % nb16;
  if (k >= K) return;
  const float* g = w + (size_t)k * N + nb * 16;
  __bf16* dst = o + ((size_t)(k >> 5) * nb16 + nb) * 512 + ((k >> 4) & 1) * 256 + (k & 15);
#pragma unroll
  for (int i = 0; i < 16; ++i) dst[i * 16] = (__bf16)g[i];
}

// ---------------------------------------------------------------------------
// GEMM: C(f32, MxN) = A(bf16 natural, MxK) * Bsw(bf16 swizzled, KxN)
// 128x128 tile / 256 threads / double-buffered async LDS staging.
// Wave w owns rows [16w,16w+16) x 128 cols (8 C frags).
// ---------------------------------------------------------------------------
__global__ __launch_bounds__(256)
void gemm_wmma_bf16(const __bf16* __restrict__ A, const __bf16* __restrict__ Bsw,
                    float* __restrict__ C, int M, int N, int K)
{
  __shared__ __align__(64) __bf16 As[2][4096];  // 128x32, A-frag-major (8 blocks)
  __shared__ __align__(64) __bf16 Bs[2][4096];  // 32x128, B-frag-major (8 blocks)
  const int tid  = threadIdx.x;
  const int lane = tid & 31;
  const int wave = tid >> 5;
  const int hi   = lane >> 4;
  const int ln   = lane & 15;
  const int m0   = blockIdx.y * 128;
  const int n0   = blockIdx.x * 128;
  const int nb16 = N >> 4;
  const int nkt  = K >> 5;

  v8f acc[8];
#pragma unroll
  for (int f = 0; f < 8; ++f)
#pragma unroll
    for (int j = 0; j < 8; ++j) acc[f][j] = 0.0f;

  auto copy_tile = [&](int t, int buf) {
    // A tile: 128 rows x 32 k = 512 x B128 chunks; swizzle in the LDS address.
#pragma unroll
    for (int i = 0; i < 2; ++i) {
      int c = tid * 2 + i;               // 0..511
      int m = c >> 2, kc = c & 3;        // k chunk: k0 = kc*8
      unsigned dst = ldsoff(&As[buf][(((m >> 4) * 32) + (m & 15) + 16 * (kc >> 1)) * 16
                                     + (kc & 1) * 8]);
      async_b128(dst, A + (size_t)(m0 + m) * K + t * 32 + kc * 8);
    }
    // B tile: 8 consecutive 512-elem blocks in swizzled global -> identity copy.
    const __bf16* gB = Bsw + ((size_t)t * nb16 + (n0 >> 4)) * 512;
#pragma unroll
    for (int i = 0; i < 2; ++i) {
      int c = tid * 2 + i;               // 0..511
      async_b128(ldsoff(&Bs[buf][c * 8]), gB + c * 8);
    }
  };

  copy_tile(0, 0);
  for (int t = 0; t < nkt; ++t) {
    __syncthreads();                     // all waves done computing on next buf
    if (t + 1 < nkt) { copy_tile(t + 1, (t + 1) & 1); WAIT_ASYNC4(); }
    else             { WAIT_ASYNC0(); }
    __syncthreads();                     // tile t resident for everyone
    const __bf16* Ab = As[t & 1];
    const __bf16* Bb = Bs[t & 1];
    // Preload all fragments so LDS latency pipelines under the WMMA chain.
    v16bf a = frag16(Ab + wave * 512, lane);
    v16bf bf[8];
#pragma unroll
    for (int f = 0; f < 8; ++f) bf[f] = frag16(Bb + f * 512, lane);
#pragma unroll
    for (int f = 0; f < 8; ++f)
      acc[f] = __builtin_amdgcn_wmma_f32_16x16x32_bf16(false, a, false, bf[f],
                                                       (short)0, acc[f], false, false);
  }

#pragma unroll
  for (int f = 0; f < 8; ++f)
#pragma unroll
    for (int j = 0; j < 8; ++j) {
      int m = m0 + wave * 16 + j + hi * 8;
      int n = n0 + f * 16 + ln;
      C[(size_t)m * N + n] = acc[f][j];
    }
}

// ---------------------------------------------------------------------------
// Per (token, head-slot) RMS-norm + RoPE. 24 slots/token: 16 q, 4 k, 4 v.
// q additionally pre-scaled by (1/sqrt(HEAD_DIM))*log2(e) for base-2 softmax.
// ---------------------------------------------------------------------------
__global__ __launch_bounds__(128)
void norm_rope_kernel(const float* __restrict__ qkv, const int* __restrict__ positions,
                      const float* __restrict__ qw, const float* __restrict__ kw,
                      __bf16* __restrict__ qb, __bf16* __restrict__ kb,
                      __bf16* __restrict__ vb)
{
  __shared__ float part[4];
  __shared__ float stage[128];
  const int blk = blockIdx.x;
  const int hh  = blk % 24;
  const int t   = blk / 24;
  const int d   = threadIdx.x;

  int col;
  if (hh < 16)      col = hh * 128;
  else if (hh < 20) col = 2048 + (hh - 16) * 128;
  else              col = 2560 + (hh - 20) * 128;

  float x = qkv[(size_t)t * QKV_N + col + d];

  if (hh < 20) {
    float sq = x * x;
#pragma unroll
    for (int off = 1; off < 32; off <<= 1) sq += __shfl_xor(sq, off, 32);
    if ((d & 31) == 0) part[d >> 5] = sq;
    __syncthreads();
    float mean = (part[0] + part[1] + part[2] + part[3]) * (1.0f / 128.0f);
    x = x * rsqrtf(mean + 1e-6f) * (hh < 16 ? qw[d] : kw[d]);
    stage[d] = x;
    __syncthreads();

    const float oscale = (hh < 16) ? QSCALE_LOG2E : 1.0f;
    __bf16* dst = (hh < 16)
        ? qb + ((size_t)t * N_HEADS + hh) * HEAD_DIM
        : kb + ((size_t)t * N_KV + (hh - 16)) * HEAD_DIM;
    if (d < 64) {
      float x1 = stage[d], x2 = stage[d + 64];
      float inv = __powf(10000.0f, -(float)d * (1.0f / 64.0f));
      float ang = (float)positions[t] * inv;
      float s, c;
      __sincosf(ang, &s, &c);
      dst[d]      = (__bf16)((x1 * c - x2 * s) * oscale);
      dst[d + 64] = (__bf16)((x2 * c + x1 * s) * oscale);
    }
  } else {
    vb[((size_t)t * N_KV + (hh - 20)) * HEAD_DIM + d] = (__bf16)x;
  }
}

// ---------------------------------------------------------------------------
// Flash-style causal attention. 128 q rows per WG (8 waves x 16 rows).
// Q/K async-copied into fragment-major LDS; V coalesced-read + swizzle-store;
// K/V double-buffered; P converted C->A layout through per-wave LDS.
// Scores arrive in log2 domain (scale*log2e folded into q): softmax uses exp2.
// ---------------------------------------------------------------------------
__global__ __launch_bounds__(256)
void attn_kernel(const __bf16* __restrict__ qb, const __bf16* __restrict__ kb,
                 const __bf16* __restrict__ vb, __bf16* __restrict__ attn)
{
  __shared__ __align__(64) __bf16 Qs[16384];    // 128x128, A-major (mb,kd) blocks
  __shared__ __align__(64) __bf16 Ks[2][8192];  // 64x128, B-major (kd,f)  blocks
  __shared__ __align__(64) __bf16 Vs[2][8192];  // 64x128, B-major (kc,f)  blocks
  __shared__ __align__(64) __bf16 Ps[8192];     // 128x64, A-major per-wave blocks

  const int tid  = threadIdx.x, lane = tid & 31, wave = tid >> 5;
  const int hi   = lane >> 4,   ln   = lane & 15;
  const int qt0  = blockIdx.x * 128;
  const int h    = blockIdx.y % N_HEADS;
  const int b    = blockIdx.y / N_HEADS;
  const int kvh  = h >> 2;

  auto copy_k = [&](int kt, int buf) {          // 1024 B128 chunks, async
    const int kt0 = kt * 64;
#pragma unroll
    for (int i = 0; i < 4; ++i) {
      int c = tid * 4 + i;                      // 0..1023
      int key = c >> 4, dc = c & 15;            // d0 = dc*8
      unsigned dst = ldsoff(&Ks[buf][((((dc >> 2) * 4 + (key >> 4)) * 32)
                                      + (key & 15) + 16 * (dc & 1)) * 16
                                     + ((dc >> 1) & 1) * 8]);
      async_b128(dst, kb + ((size_t)(b * SEQ + kt0 + key) * N_KV + kvh) * HEAD_DIM + dc * 8);
    }
  };
  auto copy_v = [&](int kt, int buf) {          // coalesced read, swizzle store
    const int kt0 = kt * 64;
    int key = tid >> 2, seg = tid & 3;
    const __bf16* g = vb + ((size_t)(b * SEQ + kt0 + key) * N_KV + kvh) * HEAD_DIM + seg * 32;
#pragma unroll
    for (int i = 0; i < 32; ++i) {
      int d = seg * 32 + i;
      Vs[buf][((((key >> 5) * 8 + (d >> 4)) * 32)
               + (d & 15) + 16 * ((key >> 4) & 1)) * 16 + (key & 15)] = g[i];
    }
  };

  {  // Q tile: 2048 B128 chunks, async, A-major swizzle
#pragma unroll
    for (int i = 0; i < 8; ++i) {
      int c = tid * 8 + i;                      // 0..2047
      int m = c >> 4, dc = c & 15;              // d0 = dc*8
      unsigned dst = ldsoff(&Qs[((((m >> 4) * 4 + (dc >> 2)) * 32)
                                 + (m & 15) + 16 * (dc & 1)) * 16
                                + ((dc >> 1) & 1) * 8]);
      async_b128(dst, qb + ((size_t)(b * SEQ + qt0 + m) * N_HEADS + h) * HEAD_DIM + dc * 8);
    }
  }
  copy_k(0, 0);
  copy_v(0, 0);
  WAIT_ASYNC0();
  __syncthreads();

  v16bf qa[4];
#pragma unroll
  for (int kd = 0; kd < 4; ++kd) qa[kd] = frag16(Qs + (wave * 4 + kd) * 512, lane);

  v8f acc[8];
  float mrun[8], lrun[8];
#pragma unroll
  for (int f = 0; f < 8; ++f)
#pragma unroll
    for (int j = 0; j < 8; ++j) acc[f][j] = 0.0f;
#pragma unroll
  for (int j = 0; j < 8; ++j) { mrun[j] = -3.0e38f; lrun[j] = 0.0f; }

  const int qrow = qt0 + wave * 16 + hi * 8;
  const int nkt  = (qt0 + 128) / 64;

  for (int kt = 0; kt < nkt; ++kt) {
    if (kt + 1 < nkt) {                        // prefetch next K/V tile
      copy_k(kt + 1, (kt + 1) & 1);
      copy_v(kt + 1, (kt + 1) & 1);
      WAIT_ASYNC4();
    } else {
      WAIT_ASYNC0();
    }
    __syncthreads();                           // tile kt resident everywhere
    const __bf16* Kb = Ks[kt & 1];
    const __bf16* Vb = Vs[kt & 1];
    const int kt0 = kt * 64;

    // S = Q * K^T (log2-domain scores)
    v8f sc[4];
#pragma unroll
    for (int f = 0; f < 4; ++f)
#pragma unroll
      for (int j = 0; j < 8; ++j) sc[f][j] = 0.0f;
#pragma unroll
    for (int kd = 0; kd < 4; ++kd) {
      v16bf kf[4];
#pragma unroll
      for (int f = 0; f < 4; ++f) kf[f] = frag16(Kb + (kd * 4 + f) * 512, lane);
#pragma unroll
      for (int f = 0; f < 4; ++f)
        sc[f] = __builtin_amdgcn_wmma_f32_16x16x32_bf16(false, qa[kd], false, kf[f],
                                                        (short)0, sc[f], false, false);
    }

    // online softmax (base-2)
    float alpha[8];
#pragma unroll
    for (int j = 0; j < 8; ++j) {
      float mx = -3.0e38f;
#pragma unroll
      for (int f = 0; f < 4; ++f) {
        float s = sc[f][j];
        bool masked = (kt0 + f * 16 + ln) > (qrow + j);
        s = masked ? -3.0e38f : s;
        sc[f][j] = s;
        mx = fmaxf(mx, s);
      }
#pragma unroll
      for (int off = 1; off < 16; off <<= 1) mx = fmaxf(mx, __shfl_xor(mx, off, 32));
      float mnew = fmaxf(mrun[j], mx);
      alpha[j] = exp2f(mrun[j] - mnew);
      mrun[j] = mnew;
      float r = 0.0f;
#pragma unroll
      for (int f = 0; f < 4; ++f) {
        float s = sc[f][j];
        float p = (s <= -1.0e37f) ? 0.0f : exp2f(s - mnew);
        sc[f][j] = p;
        r += p;
      }
#pragma unroll
      for (int off = 1; off < 16; off <<= 1) r += __shfl_xor(r, off, 32);
      lrun[j] = lrun[j] * alpha[j] + r;
    }
#pragma unroll
    for (int f = 0; f < 8; ++f)
#pragma unroll
      for (int j = 0; j < 8; ++j) acc[f][j] *= alpha[j];

    // P: C-layout -> A-major LDS (per-wave private block; no barrier needed)
    __bf16* Pw = Ps + wave * 1024;
#pragma unroll
    for (int f = 0; f < 4; ++f)
#pragma unroll
      for (int j = 0; j < 8; ++j)
        Pw[(((f >> 1) * 32) + (j + hi * 8) + 16 * (f & 1)) * 16
           + ln] = (__bf16)sc[f][j];

    // O += P * V
#pragma unroll
    for (int kc = 0; kc < 2; ++kc) {
      v16bf pa = frag16(Pw + kc * 512, lane);
      v16bf vf[8];
#pragma unroll
      for (int f = 0; f < 8; ++f) vf[f] = frag16(Vb + (kc * 8 + f) * 512, lane);
#pragma unroll
      for (int f = 0; f < 8; ++f)
        acc[f] = __builtin_amdgcn_wmma_f32_16x16x32_bf16(false, pa, false, vf[f],
                                                         (short)0, acc[f], false, false);
    }
    __syncthreads();                           // all waves done before next overwrite
  }

#pragma unroll
  for (int j = 0; j < 8; ++j) {
    float inv = 1.0f / lrun[j];
#pragma unroll
    for (int f = 0; f < 8; ++f) {
      int row = qrow + j;
      int col = h * HEAD_DIM + f * 16 + ln;
      attn[(size_t)(b * SEQ + row) * (N_HEADS * HEAD_DIM) + col] = (__bf16)(acc[f][j] * inv);
    }
  }
}

// ---------------------------------------------------------------------------
extern "C" void kernel_launch(void* const* d_in, const int* in_sizes, int n_in,
                              void* d_out, int out_size, void* d_ws, size_t ws_size,
                              hipStream_t stream)
{
  (void)in_sizes; (void)n_in; (void)out_size; (void)ws_size;
  const int*   positions = (const int*)d_in[0];
  const float* hidden    = (const float*)d_in[1];
  const float* w_qkv     = (const float*)d_in[2];
  const float* w_o       = (const float*)d_in[3];
  const float* q_norm_w  = (const float*)d_in[4];
  const float* k_norm_w  = (const float*)d_in[5];
  float* out = (float*)d_out;

  // Workspace layout (~108 MB), attn bf16 overlays the dead qkv-f32 region.
  char* ws = (char*)d_ws;
  size_t o = 0;
  __bf16* hidden_bf = (__bf16*)(ws + o); o += (size_t)TOKENS * H_SIZE * 2;     // 16 MB
  __bf16* wqkv_sw   = (__bf16*)(ws + o); o += (size_t)H_SIZE * QKV_N * 2;      // 12 MB
  __bf16* wo_sw     = (__bf16*)(ws + o); o += (size_t)H_SIZE * H_SIZE * 2;     //  8 MB
  float*  qkv       = (float*) (ws + o);
  __bf16* attnb     = (__bf16*)(ws + o); o += (size_t)TOKENS * QKV_N * 4;      // 48 MB
  __bf16* qb        = (__bf16*)(ws + o); o += (size_t)TOKENS * N_HEADS * HEAD_DIM * 2;
  __bf16* kb        = (__bf16*)(ws + o); o += (size_t)TOKENS * N_KV * HEAD_DIM * 2;
  __bf16* vb        = (__bf16*)(ws + o);

  cvt_bf16_kernel<<<(TOKENS * H_SIZE) / 256, 256, 0, stream>>>(hidden, hidden_bf,
                                                               TOKENS * H_SIZE);
  wswz_kernel<<<(H_SIZE * (QKV_N / 16)) / 256, 256, 0, stream>>>(w_qkv, wqkv_sw,
                                                                 H_SIZE, QKV_N);
  wswz_kernel<<<(H_SIZE * (H_SIZE / 16)) / 256, 256, 0, stream>>>(w_o, wo_sw,
                                                                  H_SIZE, H_SIZE);

  gemm_wmma_bf16<<<dim3(QKV_N / 128, TOKENS / 128), 256, 0, stream>>>(
      hidden_bf, wqkv_sw, qkv, TOKENS, QKV_N, H_SIZE);

  norm_rope_kernel<<<dim3(TOKENS * 24), 128, 0, stream>>>(
      qkv, positions, q_norm_w, k_norm_w, qb, kb, vb);

  attn_kernel<<<dim3(SEQ / 128, BATCH * N_HEADS), 256, 0, stream>>>(qb, kb, vb, attnb);

  gemm_wmma_bf16<<<dim3(H_SIZE / 128, TOKENS / 128), 256, 0, stream>>>(
      attnb, wo_sw, out, TOKENS, H_SIZE, H_SIZE);
}
